// RetNetBlock_21973052686682
// MI455X (gfx1250) — matrix operand order repaired and checked by reading, not compiled, not run on hardware
//
#include <hip/hip_runtime.h>

// ---------- types ----------
typedef __attribute__((ext_vector_type(16))) __bf16 v16bf;
typedef __attribute__((ext_vector_type(8)))  __bf16 v8bf;
typedef __attribute__((ext_vector_type(4)))  __bf16 v4bf;
typedef __attribute__((ext_vector_type(2)))  __bf16 v2bf;
typedef __attribute__((ext_vector_type(8)))  float  v8f;

#define DEV __device__ __forceinline__

// native bf16 converts (gfx1250 has hardware bf16 cvt)
DEV __bf16 f2bf(float f) { return (__bf16)f; }
DEV float  bf2f(__bf16 b) { return (float)b; }

DEV v16bf combine8(v8bf lo8, v8bf hi8) {
  v16bf r;
#pragma unroll
  for (int i = 0; i < 8; ++i) { r[i] = lo8[i]; r[i + 8] = hi8[i]; }
  return r;
}
DEV v8f wmma_bf16(v16bf a, v16bf b, v8f c) {
  // (neg_a, A, neg_b, B, c_mod, C, reuse_a, reuse_b)
  return __builtin_amdgcn_wmma_f32_16x16x32_bf16(false, a, false, b, (short)0, c, false, false);
}

// Async copy 32 bytes (2 x b128 per lane) global -> LDS, tracked by ASYNCcnt.
// GVS addressing: mem = SADDR(base) + VADDR(voff) + INST_OFFSET; the offset
// applies to both the global and LDS addresses, so adjacent 16B chunks land
// adjacently in LDS.
DEV void async_tile_pair(unsigned lds_off, const __bf16* base, unsigned voff) {
  asm volatile(
      "global_load_async_to_lds_b128 %0, %1, %2 offset:0\n\t"
      "global_load_async_to_lds_b128 %0, %1, %2 offset:16"
      :
      : "v"(lds_off), "v"(voff), "s"(base)
      : "memory");
}
DEV void wait_async_le4() { asm volatile("s_wait_asynccnt 0x4" ::: "memory"); }
DEV void wait_async_le0() { asm volatile("s_wait_asynccnt 0x0" ::: "memory"); }

// ---------- f32 -> bf16 conversion ----------
__global__ void f32_to_bf16_kernel(const float* __restrict__ in, __bf16* __restrict__ out, int n4) {
  int i = blockIdx.x * blockDim.x + threadIdx.x;
  if (i >= n4) return;
  float4 v = reinterpret_cast<const float4*>(in)[i];
  v4bf o; o[0] = f2bf(v.x); o[1] = f2bf(v.y); o[2] = f2bf(v.z); o[3] = f2bf(v.w);
  reinterpret_cast<v4bf*>(out)[i] = o;
}

// ---------- channel GroupNorm: stats over (D/G channels x T) per (b,g); out bf16 ----------
__global__ __launch_bounds__(256) void groupnorm_kernel(
    const float* __restrict__ x, const float* __restrict__ gamma,
    const float* __restrict__ beta, __bf16* __restrict__ out,
    int T, int D, int groups) {
  const int cpg = D / groups;                     // 64 (power of two)
  const int bg = blockIdx.x;
  const int batch = bg / groups, grp = bg % groups;
  const float* base = x + (size_t)batch * T * D + grp * cpg;
  __bf16* obase = out + (size_t)batch * T * D + grp * cpg;
  const int c  = threadIdx.x & (cpg - 1);
  const int t0 = threadIdx.x / cpg;
  const int ts = 256 / cpg;                       // 4
  float s = 0.f, q = 0.f;
  for (int t = t0; t < T; t += ts) {
    float v = base[(size_t)t * D + c];
    s += v; q += v * v;
  }
  __shared__ float rs[256], rq[256];
  rs[threadIdx.x] = s; rq[threadIdx.x] = q;
  __syncthreads();
  for (int st = 128; st > 0; st >>= 1) {
    if ((int)threadIdx.x < st) { rs[threadIdx.x] += rs[threadIdx.x + st]; rq[threadIdx.x] += rq[threadIdx.x + st]; }
    __syncthreads();
  }
  const float invn = 1.0f / (float)(cpg * T);
  const float mu = rs[0] * invn;
  const float var = rq[0] * invn - mu * mu;
  const float rstd = rsqrtf(var + 1e-5f);
  const float g = gamma[grp * cpg + c], b = beta[grp * cpg + c];
  for (int t = t0; t < T; t += ts) {
    float v = base[(size_t)t * D + c];
    obase[(size_t)t * D + c] = f2bf((v - mu) * rstd * g + b);
  }
}

// ---------- per-row L2 normalize in place (bf16 rows), one wave per row ----------
__global__ __launch_bounds__(256) void l2norm_rows_kernel(__bf16* __restrict__ p, int rows, int cols) {
  const int w = threadIdx.x >> 5, lane = threadIdx.x & 31;
  const int row = blockIdx.x * 8 + w;
  if (row >= rows) return;
  __bf16* r = p + (size_t)row * cols;
  const int nch = cols >> 3;
  float s = 0.f;
  for (int cI = lane; cI < nch; cI += 32) {
    v8bf v = reinterpret_cast<const v8bf*>(r)[cI];
#pragma unroll
    for (int i = 0; i < 8; ++i) { float f = bf2f(v[i]); s += f * f; }
  }
#pragma unroll
  for (int off = 16; off > 0; off >>= 1) s += __shfl_xor(s, off, 32);
  const float sc = 1.0f / fmaxf(sqrtf(s), 1e-12f);
  for (int cI = lane; cI < nch; cI += 32) {
    v8bf v = reinterpret_cast<const v8bf*>(r)[cI];
#pragma unroll
    for (int i = 0; i < 8; ++i) v[i] = f2bf(bf2f(v[i]) * sc);
    reinterpret_cast<v8bf*>(r)[cI] = v;
  }
}

// ---------- tiled bf16 GEMM: Out[M,N] = A[M,K] * W[N,K]^T (+epilogue) ----------
// Double-buffered LDS staged by async global->LDS copies (ASYNCcnt) so the
// copy for K-step k+1 overlaps the WMMA work on K-step k.
// EPI 0: -> bf16 ; EPI 1: +bias, SiLU -> bf16 ; EPI 2: +bias, +resid -> f32
template <int EPI>
__global__ __launch_bounds__(256) void gemm_bf16_kernel(
    const __bf16* __restrict__ A, const __bf16* __restrict__ W,
    const float* __restrict__ bias, const float* __restrict__ resid,
    void* __restrict__ Out, int M, int N, int K) {
  constexpr int LDT = 40;                         // 32 + 8 pad (keeps 16B alignment)
  __shared__ __bf16 As[2][128 * LDT];
  __shared__ __bf16 Bs[2][128 * LDT];
  const int tid = threadIdx.x;
  const int lane = tid & 31, w = tid >> 5;
  const int lo = lane & 15, hi = lane >> 4;
  const int wr = w >> 2, wc = w & 3;              // 2x4 wave grid -> 64x32 per wave
  const int m0 = blockIdx.y * 128, n0 = blockIdx.x * 128;
  const int lrow = tid >> 1, lseg = (tid & 1) << 4;

  v8f acc[4][2];
#pragma unroll
  for (int i = 0; i < 4; ++i)
#pragma unroll
    for (int j = 0; j < 2; ++j) acc[i][j] = (v8f){};

  // this thread's LDS staging slot (byte address; low 32 bits of flat = LDS offset)
  unsigned ldsA[2], ldsB[2];
  ldsA[0] = (unsigned)(uintptr_t)&As[0][lrow * LDT + lseg];
  ldsA[1] = (unsigned)(uintptr_t)&As[1][lrow * LDT + lseg];
  ldsB[0] = (unsigned)(uintptr_t)&Bs[0][lrow * LDT + lseg];
  ldsB[1] = (unsigned)(uintptr_t)&Bs[1][lrow * LDT + lseg];
  const size_t arow = (size_t)(m0 + lrow) * K + lseg;   // element offsets
  const size_t wrow = (size_t)(n0 + lrow) * K + lseg;

  // prologue: stage K-step 0 into buffer 0
  async_tile_pair(ldsA[0], A, (unsigned)(arow * 2));
  async_tile_pair(ldsB[0], W, (unsigned)(wrow * 2));

  int buf = 0;
  for (int k0 = 0; k0 < K; k0 += 32) {
    if (k0 + 32 < K) {                            // stage next K-step into other buffer
      async_tile_pair(ldsA[buf ^ 1], A, (unsigned)((arow + k0 + 32) * 2));
      async_tile_pair(ldsB[buf ^ 1], W, (unsigned)((wrow + k0 + 32) * 2));
      wait_async_le4();                           // current buffer (first 4) complete
    } else {
      wait_async_le0();
    }
    __syncthreads();

    v16bf af[4], bf[2];
#pragma unroll
    for (int mt = 0; mt < 4; ++mt) {              // A frag: row=lo, K=8*hi+{0..7,16..23}
      const __bf16* p = &As[buf][(wr * 64 + mt * 16 + lo) * LDT + hi * 8];
      af[mt] = combine8(*reinterpret_cast<const v8bf*>(p),
                        *reinterpret_cast<const v8bf*>(p + 16));
    }
#pragma unroll
    for (int nt = 0; nt < 2; ++nt) {              // B frag: col=lo, K=8*hi+{0..7,16..23}
      const __bf16* p = &Bs[buf][(wc * 32 + nt * 16 + lo) * LDT + hi * 8];
      bf[nt] = combine8(*reinterpret_cast<const v8bf*>(p),
                        *reinterpret_cast<const v8bf*>(p + 16));
    }
#pragma unroll
    for (int mt = 0; mt < 4; ++mt)
#pragma unroll
      for (int nt = 0; nt < 2; ++nt)
        acc[mt][nt] = wmma_bf16(af[mt], bf[nt], acc[mt][nt]);
    __syncthreads();                              // all reads of buf done before reuse
    buf ^= 1;
  }

#pragma unroll
  for (int mt = 0; mt < 4; ++mt)
#pragma unroll
    for (int nt = 0; nt < 2; ++nt)
#pragma unroll
      for (int r = 0; r < 8; ++r) {
        int gm = m0 + wr * 64 + mt * 16 + r + hi * 8;
        int gn = n0 + wc * 32 + nt * 16 + lo;
        float v = acc[mt][nt][r];
        if (EPI >= 1) v += bias[gn];
        if (EPI == 1) v = v / (1.0f + __expf(-v));               // SiLU
        if (EPI == 2) {
          v += resid[(size_t)gm * N + gn];
          reinterpret_cast<float*>(Out)[(size_t)gm * N + gn] = v;
        } else {
          reinterpret_cast<__bf16*>(Out)[(size_t)gm * N + gn] = f2bf(v);
        }
      }
}

// ---------- retention: Y = (decay o (Q K^T * scale)) V, causal, per (b,h,n-tile) ----------
__global__ __launch_bounds__(256) void retention_kernel(
    const __bf16* __restrict__ Q, const __bf16* __restrict__ Km,
    const __bf16* __restrict__ V, __bf16* __restrict__ Y,
    int T, int D, int H) {
  constexpr int LDT = 72;                         // 64 + 8 pad
  __shared__ __bf16 ks[64 * LDT];                 // K tile [m][d]
  __shared__ __bf16 vt[64 * LDT];                 // V tile transposed [d][m]
  __shared__ __bf16 ss[64 * LDT];                 // Q tile at start, then S tile [n][m]
  const int dh = D / H;                           // 64
  const int n0 = blockIdx.x * 64;
  const int h = blockIdx.y;
  const size_t rowbase = (size_t)blockIdx.z * T;
  const int tid = threadIdx.x, lane = tid & 31, w = tid >> 5;
  const int lo = lane & 15, hi = lane >> 4;
  const int swr = (w >> 1) * 16, swc = (w & 1) * 32;  // 4x2 wave grid: 16x32 per wave
  const int lrow = tid >> 2, lseg = (tid & 3) << 4;   // 64 rows x 4 segs(16)
  const int rp = tid >> 3, seg8 = (tid & 7) << 3;     // 32 row-pairs x 8 segs(8) for V

  const float l0 = logf(1.0f / 32.0f), l1 = logf(1.0f / 512.0f);
  const float gamma = 1.0f - __expf(l0 + (l1 - l0) * ((float)h / (float)(H - 1)));
  const float log2g = log2f(gamma);
  const float scale = rsqrtf((float)dh);

  // stage Q tile into ss via async copy (consumed into regs before ss reused for S)
  {
    unsigned ldsq = (unsigned)(uintptr_t)&ss[lrow * LDT + lseg];
    async_tile_pair(ldsq, Q, (unsigned)(((rowbase + n0 + lrow) * D + h * dh + lseg) * 2));
    wait_async_le0();
  }
  __syncthreads();
  v16bf qf[2];
#pragma unroll
  for (int kk = 0; kk < 2; ++kk) {
    const __bf16* p = &ss[(swr + lo) * LDT + kk * 32 + hi * 8];
    qf[kk] = combine8(*reinterpret_cast<const v8bf*>(p),
                      *reinterpret_cast<const v8bf*>(p + 16));
  }
  v8f yacc[2] = {{}, {}};

  const unsigned ldsk = (unsigned)(uintptr_t)&ks[lrow * LDT + lseg];

  for (int m0 = 0; m0 <= n0; m0 += 64) {
    __syncthreads();                              // prior-iter LDS reads done (first: q frags)
    // async-stage K tile [m][d]
    async_tile_pair(ldsk, Km, (unsigned)(((rowbase + m0 + lrow) * D + h * dh + lseg) * 2));
    // V tile: row-pair per thread, pack 2 m-values per dword -> transposed vt[d][m]
    {
      const __bf16* vg = V + (rowbase + m0 + 2 * rp) * D + h * dh + seg8;
      union { uint4 q; v8bf v; } r0, r1;
      r0.q = *reinterpret_cast<const uint4*>(vg);
      r1.q = *reinterpret_cast<const uint4*>(vg + D);
#pragma unroll
      for (int i = 0; i < 8; ++i) {
        v2bf t; t[0] = r0.v[i]; t[1] = r1.v[i];
        *reinterpret_cast<v2bf*>(&vt[(seg8 + i) * LDT + 2 * rp]) = t;   // ds_store_b32
      }
      if (m0 + 64 <= n0) {                        // gfx1250 global_prefetch_b8
        __builtin_prefetch(Km + (rowbase + m0 + 64 + lrow) * D + h * dh + lseg, 0, 1);
        __builtin_prefetch(V + (rowbase + m0 + 64 + 2 * rp) * D + h * dh + seg8, 0, 1);
      }
    }
    wait_async_le0();
    __syncthreads();

    // S = Q K^T  (K dim = dh = 64 -> 2 WMMA k-steps)
    v8f sacc[2] = {{}, {}};
#pragma unroll
    for (int kk = 0; kk < 2; ++kk)
#pragma unroll
      for (int nt = 0; nt < 2; ++nt) {
        const __bf16* p = &ks[(swc + nt * 16 + lo) * LDT + kk * 32 + hi * 8];
        v16bf bfrag = combine8(*reinterpret_cast<const v8bf*>(p),
                               *reinterpret_cast<const v8bf*>(p + 16));
        sacc[nt] = wmma_bf16(qf[kk], bfrag, sacc[nt]);
      }

    // decay * scale, causal mask; write S as bf16 to LDS
#pragma unroll
    for (int nt = 0; nt < 2; ++nt)
#pragma unroll
      for (int r = 0; r < 8; ++r) {
        int i = swr + r + hi * 8;
        int j = swc + nt * 16 + lo;
        int d = (n0 + i) - (m0 + j);
        float f = (d >= 0) ? scale * exp2f((float)d * log2g) : 0.0f;
        ss[i * LDT + j] = f2bf(sacc[nt][r] * f);
      }
    __syncthreads();

    // Y += S V  (K dim = 64 -> 2 k-steps), accumulate across m-tiles
#pragma unroll
    for (int kk = 0; kk < 2; ++kk) {
      const __bf16* pa = &ss[(swr + lo) * LDT + kk * 32 + hi * 8];
      v16bf afrag = combine8(*reinterpret_cast<const v8bf*>(pa),
                             *reinterpret_cast<const v8bf*>(pa + 16));
#pragma unroll
      for (int nt = 0; nt < 2; ++nt) {
        const __bf16* pb = &vt[(swc + nt * 16 + lo) * LDT + kk * 32 + hi * 8];
        v16bf bfrag = combine8(*reinterpret_cast<const v8bf*>(pb),
                               *reinterpret_cast<const v8bf*>(pb + 16));
        yacc[nt] = wmma_bf16(afrag, bfrag, yacc[nt]);
      }
    }
  }

#pragma unroll
  for (int nt = 0; nt < 2; ++nt)
#pragma unroll
    for (int r = 0; r < 8; ++r) {
      int i = swr + r + hi * 8;
      int j = swc + nt * 16 + lo;
      Y[(rowbase + n0 + i) * D + h * dh + j] = f2bf(yacc[nt][r]);
    }
}

// ---------- launch ----------
extern "C" void kernel_launch(void* const* d_in, const int* in_sizes, int n_in,
                              void* d_out, int out_size, void* d_ws, size_t ws_size,
                              hipStream_t stream) {
  (void)in_sizes; (void)n_in; (void)out_size; (void)ws_size;
  const int B = 4, T = 1024, D = 1024, H = 16;
  const int M = B * T;        // 4096
  const int D2 = 2 * D;       // 2048

  const float* x  = (const float*)d_in[0];
  const float* ng = (const float*)d_in[1];
  const float* nb = (const float*)d_in[2];
  const float* qw = (const float*)d_in[3];
  const float* kw = (const float*)d_in[4];
  const float* vw = (const float*)d_in[5];
  const float* ow = (const float*)d_in[6];
  const float* ob = (const float*)d_in[7];
  const float* fg = (const float*)d_in[8];
  const float* fb = (const float*)d_in[9];
  const float* w1 = (const float*)d_in[10];
  const float* b1 = (const float*)d_in[11];
  const float* w2 = (const float*)d_in[12];
  const float* b2 = (const float*)d_in[13];

  char* ws = (char*)d_ws;
  size_t off = 0;
  auto take = [&](size_t bytes) -> char* {
    char* p = ws + off;
    off += (bytes + 255) & ~(size_t)255;
    return p;
  };
  __bf16* qw_b = (__bf16*)take((size_t)D * D * 2);
  __bf16* kw_b = (__bf16*)take((size_t)D * D * 2);
  __bf16* vw_b = (__bf16*)take((size_t)D * D * 2);
  __bf16* ow_b = (__bf16*)take((size_t)D * D * 2);
  __bf16* w1_b = (__bf16*)take((size_t)D2 * D * 2);
  __bf16* w2_b = (__bf16*)take((size_t)D * D2 * 2);
  __bf16* xn_b = (__bf16*)take((size_t)M * D * 2);   // reused for 2nd groupnorm
  __bf16* q_b  = (__bf16*)take((size_t)M * D * 2);
  __bf16* k_b  = (__bf16*)take((size_t)M * D * 2);
  __bf16* v_b  = (__bf16*)take((size_t)M * D * 2);
  __bf16* y_b  = (__bf16*)take((size_t)M * D * 2);
  float*  x1_f = (float*)take((size_t)M * D * 4);
  __bf16* h_b  = (__bf16*)take((size_t)M * D2 * 2);

  auto cvt = [&](const float* src, __bf16* dst, size_t n) {
    int n4 = (int)(n / 4);
    f32_to_bf16_kernel<<<(n4 + 255) / 256, 256, 0, stream>>>(src, dst, n4);
  };
  cvt(qw, qw_b, (size_t)D * D);
  cvt(kw, kw_b, (size_t)D * D);
  cvt(vw, vw_b, (size_t)D * D);
  cvt(ow, ow_b, (size_t)D * D);
  cvt(w1, w1_b, (size_t)D2 * D);
  cvt(w2, w2_b, (size_t)D * D2);

  // retention sub-block
  groupnorm_kernel<<<dim3(B * H), 256, 0, stream>>>(x, ng, nb, xn_b, T, D, H);
  gemm_bf16_kernel<0><<<dim3(D / 128, M / 128), 256, 0, stream>>>(xn_b, qw_b, nullptr, nullptr, q_b, M, D, D);
  gemm_bf16_kernel<0><<<dim3(D / 128, M / 128), 256, 0, stream>>>(xn_b, kw_b, nullptr, nullptr, k_b, M, D, D);
  gemm_bf16_kernel<0><<<dim3(D / 128, M / 128), 256, 0, stream>>>(xn_b, vw_b, nullptr, nullptr, v_b, M, D, D);
  l2norm_rows_kernel<<<M / 8, 256, 0, stream>>>(q_b, M, D);
  l2norm_rows_kernel<<<M / 8, 256, 0, stream>>>(k_b, M, D);
  retention_kernel<<<dim3(T / 64, H, B), 256, 0, stream>>>(q_b, k_b, v_b, y_b, T, D, H);
  gemm_bf16_kernel<2><<<dim3(D / 128, M / 128), 256, 0, stream>>>(y_b, ow_b, ob, x, x1_f, M, D, D);

  // FFN sub-block
  groupnorm_kernel<<<dim3(B * H), 256, 0, stream>>>(x1_f, fg, fb, xn_b, T, D, H);
  gemm_bf16_kernel<1><<<dim3(D2 / 128, M / 128), 256, 0, stream>>>(xn_b, w1_b, b1, nullptr, h_b, M, D2, D);
  gemm_bf16_kernel<2><<<dim3(D / 128, M / 128), 256, 0, stream>>>(h_b, w2_b, b2, x1_f, (float*)d_out, M, D, D2);
}
